// DoubleGSOFTCrossAttnProcessor_36180804501934
// MI455X (gfx1250) — matrix operand order repaired and checked by reading, not compiled, not run on hardware
//
#include <hip/hip_runtime.h>

#define HID 1280
#define CROSS 768
#define NBLK 16
#define HEADS 8
#define HEAD_DIM 160
#define BATCH 8
#define SQ 4096
#define SK 77
#define SKP 80    // keys padded for score WMMA (5 n-tiles)
#define SKP2 96   // keys padded to K-chunk multiple for probs@V (3 k-chunks)
#define BH 80
#define BC 48

typedef __bf16 bf16;
typedef __bf16 v16bf __attribute__((ext_vector_type(16)));
typedef float  v8f   __attribute__((ext_vector_type(8)));
typedef uint32_t u32x4 __attribute__((ext_vector_type(4)));
typedef uint32_t u32x8 __attribute__((ext_vector_type(8)));

// Load a WMMA fragment as two aligned 16-byte chunks (p0: elements 0-7,
// p1: elements 8-15). Layouts are arranged so both chunks are contiguous runs.
static __device__ __forceinline__ v16bf frag_ld2(const bf16* p0, const bf16* p1) {
  union { v16bf v; uint4 q[2]; } u;
  u.q[0] = *(const uint4*)p0;
  u.q[1] = *(const uint4*)p1;
  return u.v;
}

// ---------------------------------------------------------------------------
// TDM 2D tile load: global (strided rows) -> LDS (contiguous rows, optional
// HW padding). D# group0 (4 SGPRs) + group1 (8 SGPRs) per ISA 8.3/8.4;
// tensor_load_to_lds 2-group form (VADDR2/3 = NULL, tensors <= 2D).
// ---------------------------------------------------------------------------
static __device__ __forceinline__ void tdm_load_2d(
    uint32_t lds_off, const void* gptr,
    uint32_t tensor_d0, uint32_t tensor_d1,
    uint32_t tile_d0, uint32_t tile_d1,
    uint32_t stride_elems, uint32_t g1dw0) {
  uint64_t ga = (uint64_t)(uintptr_t)gptr;
  u32x4 g0;
  g0[0] = 1u;                                           // count=1, user D#
  g0[1] = lds_off;                                      // lds_addr (bytes)
  g0[2] = (uint32_t)ga;                                 // global_addr[31:0]
  g0[3] = (uint32_t)((ga >> 32) & 0x1FFFFFFu) | (2u << 30);  // [56:32] | type=2
  u32x8 g1;
  g1[0] = g1dw0;                                        // mask=0, data_size, pad cfg
  g1[1] = (tensor_d0 & 0xFFFFu) << 16;                  // tensor_dim0 lo
  g1[2] = (tensor_d0 >> 16) | ((tensor_d1 & 0xFFFFu) << 16);
  g1[3] = (tensor_d1 >> 16) | (tile_d0 << 16);
  g1[4] = tile_d1;                                      // tile_dim2 = 0
  g1[5] = stride_elems;                                 // tensor_dim0_stride lo32
  g1[6] = 0u;
  g1[7] = 0u;
  asm volatile("tensor_load_to_lds %0, %1" :: "s"(g0), "s"(g1) : "memory");
}

#define TDM_DW0_NOPAD   (0x00010000u)                               // 2B elems
#define TDM_DW0_PAD(iv, am) (0x00010000u | (1u << 20) | ((iv) << 22) | ((am) << 25))

// ---------------------------------------------------------------------------
// 1) Cayley map: Q = (I - A)^-1 (I + A), A = P - P^T. Gauss-Jordan in LDS.
// ---------------------------------------------------------------------------
__global__ __launch_bounds__(256) void cayley_kernel(const float* __restrict__ P,
                                                     int n, float* __restrict__ Q) {
  __shared__ float Ms[80 * 81];
  __shared__ float Rs[80 * 81];
  __shared__ float fcol[80];
  const int ST = 81;
  const int g = blockIdx.x;
  const float* Pg = P + g * n * n;
  const int tid = threadIdx.x;

  for (int idx = tid; idx < n * n; idx += blockDim.x) {
    int i = idx / n, j = idx % n;
    float a = Pg[i * n + j] - Pg[j * n + i];
    float id = (i == j) ? 1.f : 0.f;
    Ms[i * ST + j] = id - a;
    Rs[i * ST + j] = id + a;
  }
  for (int k = 0; k < n; ++k) {
    __syncthreads();
    float piv = Ms[k * ST + k];
    for (int i = tid; i < n; i += blockDim.x) fcol[i] = Ms[i * ST + k];
    __syncthreads();
    float inv = 1.f / piv;
    for (int j = tid; j < n; j += blockDim.x) {
      Ms[k * ST + j] *= inv;
      Rs[k * ST + j] *= inv;
    }
    __syncthreads();
    for (int idx = tid; idx < n * n; idx += blockDim.x) {
      int i = idx / n, j = idx % n;
      if (i != k) {
        float f = fcol[i];
        Ms[i * ST + j] -= f * Ms[k * ST + j];
        Rs[i * ST + j] -= f * Rs[k * ST + j];
      }
    }
  }
  __syncthreads();
  for (int idx = tid; idx < n * n; idx += blockDim.x) {
    int i = idx / n, j = idx % n;
    Q[g * n * n + i * n + j] = Rs[i * ST + j];
  }
}

// ---------------------------------------------------------------------------
// 2a) T = (W^T @ BD(Qout)) * diag(scale)
// ---------------------------------------------------------------------------
__global__ void fuse_a_kernel(const float* __restrict__ W, const float* __restrict__ Qout,
                              const float* __restrict__ scale, int in_dim,
                              float* __restrict__ T) {
  int idx = blockIdx.x * blockDim.x + threadIdx.x;
  if (idx >= in_dim * HID) return;
  int i = idx / HID, c = idx % HID;
  int g = c / BH, cc = c % BH;
  const float* Qg = Qout + g * BH * BH;
  const float* Wb = W + (size_t)(g * BH) * in_dim + i;
  float acc = 0.f;
#pragma unroll 4
  for (int j = 0; j < BH; ++j) acc += Wb[(size_t)j * in_dim] * Qg[j * BH + cc];
  T[idx] = acc * scale[c];
}

// 2b) WfT = (BD(Qin) @ T)^T  -- stored TRANSPOSED [N=HID][K=in_dim] bf16 so
//     GEMM B-fragments are contiguous along K.
__global__ void fuse_b_kernel(const float* __restrict__ Qin, int bs,
                              const float* __restrict__ T, int in_dim,
                              bf16* __restrict__ WfT) {
  int idx = blockIdx.x * blockDim.x + threadIdx.x;
  if (idx >= in_dim * HID) return;
  int r = idx / HID, c = idx % HID;
  int g = r / bs, rr = r % bs;
  const float* Qg = Qin + g * bs * bs + rr * bs;
  const float* Tb = T + (size_t)(g * bs) * HID + c;
  float acc = 0.f;
#pragma unroll 4
  for (int j = 0; j < bs; ++j) acc += Qg[j] * Tb[(size_t)j * HID];
  WfT[(size_t)c * in_dim + r] = (bf16)acc;
}

// ---------------------------------------------------------------------------
// 3) f32 -> bf16 conversion
// ---------------------------------------------------------------------------
__global__ void f32_to_bf16_kernel(const float* __restrict__ x, bf16* __restrict__ y, int n) {
  int i = blockIdx.x * blockDim.x + threadIdx.x;
  if (i < n) y[i] = (bf16)x[i];
}

// ---------------------------------------------------------------------------
// 4) Tiled WMMA GEMM: C[M,N] = A[M,K] @ Bt[N,K]^T, bf16 in, f32 accumulate.
//    128x64 block tile, 8 waves x (16 rows x 64 cols), K-step 32.
//    Both tiles staged by the TDM with HW row padding -> LDS stride 40 halves;
//    every fragment is two 16-byte LDS loads.
// ---------------------------------------------------------------------------
template <int OUT_F32>
__global__ __launch_bounds__(256) void gemm_bf16_kernel(
    const bf16* __restrict__ A, const bf16* __restrict__ Bt,
    void* __restrict__ Cout, const float* __restrict__ bias,
    int M, int N, int K) {
  constexpr int BM = 128, BN = 64, BK = 32;
  constexpr int LST = 40;                 // 32-half row + 8-half TDM pad
  __shared__ bf16 As[BM * LST];           // [m][k]
  __shared__ bf16 Bs[BN * LST];           // [n][k]

  const int tid = threadIdx.x;
  const int lane = tid & 31;
  const int w = tid >> 5;
  const int lane_lo = (lane < 16) ? 1 : 0;
  const int lrow = lane & 15;
  const int m0 = blockIdx.y * BM;
  const int n0 = blockIdx.x * BN;
  const int klo = lane_lo ? 0 : 8;        // A-fragment chunk base
  const int kb0 = lane_lo ? 0 : 16;       // unused; B fragments also use klo

  v8f acc[4] = {};

  const uint32_t asOff = (uint32_t)(uintptr_t)&As[0];
  const uint32_t bsOff = (uint32_t)(uintptr_t)&Bs[0];
  const uint32_t mrem = (uint32_t)(M - m0);   // TDM zero-fills OOB rows
  const uint32_t nrem = (uint32_t)(N - n0);

  for (int k0 = 0; k0 < K; k0 += BK) {
    if (w == 0) {
      // A tile: BM rows x BK, row stride K; pad every 16 dw by 4 dw
      tdm_load_2d(asOff, A + (size_t)m0 * K + k0,
                  (uint32_t)K, mrem, BK, BM, (uint32_t)K, TDM_DW0_PAD(3u, 3u));
      // B tile: BN rows x BK from Bt[N][K]; same pad config
      tdm_load_2d(bsOff, Bt + (size_t)n0 * K + k0,
                  (uint32_t)K, nrem, BK, BN, (uint32_t)K, TDM_DW0_PAD(3u, 3u));
      __builtin_amdgcn_s_wait_tensorcnt(0);
    }
    __syncthreads();

    const bf16* pa = &As[((w << 4) + lrow) * LST + klo];
    v16bf a = frag_ld2(pa, pa + 16);
#pragma unroll
    for (int nt = 0; nt < 4; ++nt) {
      const bf16* pb = &Bs[((nt << 4) + lrow) * LST + klo];
      v16bf bfrag = frag_ld2(pb, pb + 16);
      acc[nt] = __builtin_amdgcn_wmma_f32_16x16x32_bf16(
          false, a, false, bfrag, (short)0, acc[nt], false, false);
    }
    __syncthreads();
  }

#pragma unroll
  for (int nt = 0; nt < 4; ++nt) {
#pragma unroll
    for (int v = 0; v < 8; ++v) {
      int r = m0 + (w << 4) + v + (lane_lo ? 0 : 8);
      int c = n0 + (nt << 4) + lrow;
      if (r < M) {
        if (OUT_F32)
          ((float*)Cout)[(size_t)r * N + c] = acc[nt][v] + bias[c];
        else
          ((bf16*)Cout)[(size_t)r * N + c] = (bf16)acc[nt][v];
      }
    }
  }
}

// ---------------------------------------------------------------------------
// 5) Cross-attention. K tile staged row-major via TDM (feeds score B-frags
//    contiguously); V gathered transposed [d][s]; probs LDS aliases sK.
// ---------------------------------------------------------------------------
__global__ __launch_bounds__(256) void attn_kernel(
    const bf16* __restrict__ qb, const bf16* __restrict__ kb,
    const bf16* __restrict__ vb, bf16* __restrict__ ob) {
  __shared__ bf16 sK[SKP * HEAD_DIM];     // [s][d]  25.6 KB (reused for probs)
  __shared__ bf16 sVt[HEAD_DIM * SKP2];   // [d][s]  30.7 KB

  const int tid = threadIdx.x;
  const int lane = tid & 31;
  const int w = tid >> 5;
  const int lane_lo = (lane < 16) ? 1 : 0;
  const int lrow = lane & 15;
  const int b = blockIdx.y / HEADS;
  const int h = blockIdx.y % HEADS;
  const int q0 = blockIdx.x * 128;
  const int klo = lane_lo ? 0 : 8;
  const int kb0 = lane_lo ? 0 : 16;
  const float scl = 0.07905694150420949f;  // 160^-0.5

  // K rows 0..76 via TDM: [SK][HEAD_DIM] tile, row stride HID, contiguous LDS
  if (w == 0) {
    tdm_load_2d((uint32_t)(uintptr_t)&sK[0],
                kb + (size_t)(b * SK) * HID + h * HEAD_DIM,
                HID, SK, HEAD_DIM, SK, HID, TDM_DW0_NOPAD);
    __builtin_amdgcn_s_wait_tensorcnt(0);
  }
  // zero K pad rows 77..79 (disjoint from TDM region)
  for (int idx = tid; idx < (SKP - SK) * HEAD_DIM; idx += 256)
    sK[SK * HEAD_DIM + idx] = (bf16)0.f;
  // V transposed gather: sVt[d][s], zero s >= 77
  for (int idx = tid; idx < HEAD_DIM * SKP2; idx += 256) {
    int d = idx / SKP2, s = idx % SKP2;
    sVt[idx] = (s < SK) ? vb[(size_t)(b * SK + s) * HID + h * HEAD_DIM + d] : (bf16)0.f;
  }
  __syncthreads();

  // ---- scores = Q @ K^T : 5 K-chunks of 32 over head_dim, 5 n-tiles ----
  v8f accS[5] = {};
  const int qrow = q0 + (w << 4) + lrow;
  const bf16* qrp = qb + (size_t)(b * SQ + qrow) * HID + h * HEAD_DIM;
  for (int kc = 0; kc < 5; ++kc) {
    const bf16* pa = qrp + kc * 32 + klo;
    v16bf a = frag_ld2(pa, pa + 16);          // 2x global b128
#pragma unroll
    for (int nt = 0; nt < 5; ++nt) {
      const bf16* pb = &sK[((nt << 4) + lrow) * HEAD_DIM + kc * 32 + klo];
      v16bf bfrag = frag_ld2(pb, pb + 16);
      accS[nt] = __builtin_amdgcn_wmma_f32_16x16x32_bf16(
          false, a, false, bfrag, (short)0, accS[nt], false, false);
    }
  }
  __syncthreads();          // all waves done reading sK
  bf16* sP = sK;            // alias: [128 rows][96] = 12288 halves <= 12800

  // ---- softmax over 77 keys ----
#pragma unroll
  for (int v = 0; v < 8; ++v) {
    float vals[5];
    float mx = -3.0e38f;
#pragma unroll
    for (int nt = 0; nt < 5; ++nt) {
      int col = (nt << 4) + lrow;
      float val = accS[nt][v] * scl;
      vals[nt] = (col < SK) ? val : -3.0e38f;
      mx = fmaxf(mx, vals[nt]);
    }
#pragma unroll
    for (int off = 8; off >= 1; off >>= 1) mx = fmaxf(mx, __shfl_xor(mx, off, 32));
    float sum = 0.f;
#pragma unroll
    for (int nt = 0; nt < 5; ++nt) {
      int col = (nt << 4) + lrow;
      float p = (col < SK) ? __expf(vals[nt] - mx) : 0.f;
      vals[nt] = p;
      sum += p;
    }
#pragma unroll
    for (int off = 8; off >= 1; off >>= 1) sum += __shfl_xor(sum, off, 32);
    float rinv = 1.f / sum;
    int prow = (w << 4) + v + (lane_lo ? 0 : 8);
#pragma unroll
    for (int nt = 0; nt < 5; ++nt)
      sP[prow * SKP2 + (nt << 4) + lrow] = (bf16)(vals[nt] * rinv);
    sP[prow * SKP2 + 80 + lrow] = (bf16)0.f;   // zero K-padding 80..95
  }

  // ---- out = probs @ V : K=96 (3 chunks), N=160 (10 n-tiles) ----
  v8f accO[10] = {};
  const bf16* sPw = sP + (size_t)(w << 4) * SKP2;
  for (int kc = 0; kc < 3; ++kc) {
    const bf16* pa = &sPw[lrow * SKP2 + kc * 32 + klo];
    v16bf a = frag_ld2(pa, pa + 16);
#pragma unroll
    for (int nt = 0; nt < 10; ++nt) {
      const bf16* pb = &sVt[((nt << 4) + lrow) * SKP2 + kc * 32 + klo];
      v16bf bfrag = frag_ld2(pb, pb + 16);
      accO[nt] = __builtin_amdgcn_wmma_f32_16x16x32_bf16(
          false, a, false, bfrag, (short)0, accO[nt], false, false);
    }
  }
#pragma unroll
  for (int nt = 0; nt < 10; ++nt) {
#pragma unroll
    for (int v = 0; v < 8; ++v) {
      int r = q0 + (w << 4) + v + (lane_lo ? 0 : 8);
      int c = h * HEAD_DIM + (nt << 4) + lrow;
      ob[(size_t)(b * SQ + r) * HID + c] = (bf16)accO[nt][v];
    }
  }
}

// ---------------------------------------------------------------------------
// Host-side orchestration
// ---------------------------------------------------------------------------
extern "C" void kernel_launch(void* const* d_in, const int* in_sizes, int n_in,
                              void* d_out, int out_size, void* d_ws, size_t ws_size,
                              hipStream_t stream) {
  (void)in_sizes; (void)n_in; (void)out_size; (void)ws_size;
  const float* hs      = (const float*)d_in[0];
  const float* enc     = (const float*)d_in[1];
  const float* Pq_in   = (const float*)d_in[2];
  const float* Pk_in   = (const float*)d_in[3];
  const float* Pv_in   = (const float*)d_in[4];
  const float* Pout_in = (const float*)d_in[5];
  const float* Pq_out  = (const float*)d_in[6];
  const float* Pk_out  = (const float*)d_in[7];
  const float* Pv_out  = (const float*)d_in[8];
  const float* Pout_out= (const float*)d_in[9];
  const float* Wq      = (const float*)d_in[10];
  const float* Wk      = (const float*)d_in[11];
  const float* Wv      = (const float*)d_in[12];
  const float* Wout    = (const float*)d_in[13];
  const float* bout    = (const float*)d_in[14];
  const float* q_scale = (const float*)d_in[15];
  const float* k_scale = (const float*)d_in[16];
  const float* v_scale = (const float*)d_in[17];
  const float* out_scale = (const float*)d_in[18];

  char* wsp = (char*)d_ws;
  auto carve = [&](size_t bytes) -> void* {
    void* p = (void*)wsp;
    wsp += (bytes + 255) & ~(size_t)255;
    return p;
  };
  float* Qqi = (float*)carve((size_t)NBLK * BH * BH * 4);
  float* Qki = (float*)carve((size_t)NBLK * BC * BC * 4);
  float* Qvi = (float*)carve((size_t)NBLK * BC * BC * 4);
  float* Qoi = (float*)carve((size_t)NBLK * BH * BH * 4);
  float* Qqo = (float*)carve((size_t)NBLK * BH * BH * 4);
  float* Qko = (float*)carve((size_t)NBLK * BH * BH * 4);
  float* Qvo = (float*)carve((size_t)NBLK * BH * BH * 4);
  float* Qoo = (float*)carve((size_t)NBLK * BH * BH * 4);
  float* Tf  = (float*)carve((size_t)HID * HID * 4);
  bf16* Wqf  = (bf16*)carve((size_t)HID * HID * 2);     // transposed [N][K]
  bf16* Wkf  = (bf16*)carve((size_t)HID * CROSS * 2);   // transposed [N][K]
  bf16* Wvf  = (bf16*)carve((size_t)HID * CROSS * 2);   // transposed [N][K]
  bf16* Wof  = (bf16*)carve((size_t)HID * HID * 2);     // transposed [N][K]
  bf16* xbf  = (bf16*)carve((size_t)BATCH * SQ * HID * 2);
  bf16* ebf  = (bf16*)carve((size_t)BATCH * SK * CROSS * 2);
  bf16* qbf  = (bf16*)carve((size_t)BATCH * SQ * HID * 2);
  bf16* kbf  = (bf16*)carve((size_t)BATCH * SK * HID * 2);
  bf16* vbf  = (bf16*)carve((size_t)BATCH * SK * HID * 2);
  bf16* abf  = (bf16*)carve((size_t)BATCH * SQ * HID * 2);

  // 1) Cayley maps
  cayley_kernel<<<NBLK, 256, 0, stream>>>(Pq_in,   BH, Qqi);
  cayley_kernel<<<NBLK, 256, 0, stream>>>(Pk_in,   BC, Qki);
  cayley_kernel<<<NBLK, 256, 0, stream>>>(Pv_in,   BC, Qvi);
  cayley_kernel<<<NBLK, 256, 0, stream>>>(Pout_in, BH, Qoi);
  cayley_kernel<<<NBLK, 256, 0, stream>>>(Pq_out,  BH, Qqo);
  cayley_kernel<<<NBLK, 256, 0, stream>>>(Pk_out,  BH, Qko);
  cayley_kernel<<<NBLK, 256, 0, stream>>>(Pv_out,  BH, Qvo);
  cayley_kernel<<<NBLK, 256, 0, stream>>>(Pout_out,BH, Qoo);

  // 2) Fused weights (scale folded into columns; stored transposed)
  int nqq = HID * HID, nkk = CROSS * HID;
  fuse_a_kernel<<<(nqq + 255) / 256, 256, 0, stream>>>(Wq, Qqo, q_scale, HID, Tf);
  fuse_b_kernel<<<(nqq + 255) / 256, 256, 0, stream>>>(Qqi, BH, Tf, HID, Wqf);
  fuse_a_kernel<<<(nkk + 255) / 256, 256, 0, stream>>>(Wk, Qko, k_scale, CROSS, Tf);
  fuse_b_kernel<<<(nkk + 255) / 256, 256, 0, stream>>>(Qki, BC, Tf, CROSS, Wkf);
  fuse_a_kernel<<<(nkk + 255) / 256, 256, 0, stream>>>(Wv, Qvo, v_scale, CROSS, Tf);
  fuse_b_kernel<<<(nkk + 255) / 256, 256, 0, stream>>>(Qvi, BC, Tf, CROSS, Wvf);
  fuse_a_kernel<<<(nqq + 255) / 256, 256, 0, stream>>>(Wout, Qoo, out_scale, HID, Tf);
  fuse_b_kernel<<<(nqq + 255) / 256, 256, 0, stream>>>(Qoi, BH, Tf, HID, Wof);

  // 3) Activation conversion
  int nx = BATCH * SQ * HID, ne = BATCH * SK * CROSS;
  f32_to_bf16_kernel<<<(nx + 255) / 256, 256, 0, stream>>>(hs, xbf, nx);
  f32_to_bf16_kernel<<<(ne + 255) / 256, 256, 0, stream>>>(enc, ebf, ne);

  // 4) Projections (WMMA + TDM staging)
  dim3 gq(HID / 64, (BATCH * SQ) / 128);
  gemm_bf16_kernel<0><<<gq, 256, 0, stream>>>(xbf, Wqf, qbf, nullptr, BATCH * SQ, HID, HID);
  dim3 gk(HID / 64, (BATCH * SK + 127) / 128);
  gemm_bf16_kernel<0><<<gk, 256, 0, stream>>>(ebf, Wkf, kbf, nullptr, BATCH * SK, HID, CROSS);
  gemm_bf16_kernel<0><<<gk, 256, 0, stream>>>(ebf, Wvf, vbf, nullptr, BATCH * SK, HID, CROSS);

  // 5) Attention (WMMA + TDM staging for K)
  attn_kernel<<<dim3(SQ / 128, BATCH * HEADS), 256, 0, stream>>>(qbf, kbf, vbf, abf);

  // 6) Output projection with bias, f32 out (WMMA)
  gemm_bf16_kernel<1><<<gq, 256, 0, stream>>>(abf, Wof, d_out, bout, BATCH * SQ, HID, HID);
}
